// QueryFormer_32615981646544
// MI455X (gfx1250) — compile-verified, matrix operand order checked by reading, not run on hardware
//
#include <hip/hip_runtime.h>
#include <hip/hip_bf16.h>
#include <math.h>

// ---------------------------------------------------------------------------
// QueryFormer forward for MI455X (gfx1250, wave32, WMMA).
// Memory-bound model (~10 GB activation traffic vs 23.3 TB/s HBM; weights fit
// in 192MB L2 as bf16). All GEMMs use v_wmma_f32_16x16x32_bf16 with fused
// bias/activation/residual epilogues; GEMM tiles are streamed into LDS with
// CDNA5 GLOBAL_LOAD_ASYNC_TO_LDS_B128 (ASYNCcnt) using double buffering so
// copies overlap WMMA; attention is flash-style per (b,h,qtile) with the
// tree-bias (attn_bias + rel_pos_emb gather + svd) fused into the score
// epilogue so the 134MB [B,H,S,S] bias/score tensors are never stored.
// ---------------------------------------------------------------------------

typedef __attribute__((ext_vector_type(16))) __bf16 v16bf;
typedef __attribute__((ext_vector_type(8)))  __bf16 v8bf;
typedef __attribute__((ext_vector_type(8)))  float  v8f;

#define BM 128
#define BN 128
#define BK 32
#define LDSK 40                      // padded LDS row stride (bf16), 16B aligned
#define SCALE_DK 0.12403473458920847f // 65^-0.5

__device__ __forceinline__ v8f wmma_bf16(v16bf a, v16bf b, v8f c) {
  return __builtin_amdgcn_wmma_f32_16x16x32_bf16(false, a, false, b,
                                                 (short)0, c, false, false);
}

// CDNA5 async global->LDS copy, 16B per lane, tracked by ASYNCcnt.
// LDS byte offset = low 32 bits of the generic pointer (ISA 10.2).
__device__ __forceinline__ void async_copy_b128(void* lds_dst, const void* gsrc) {
  unsigned int loff = (unsigned int)(unsigned long long)lds_dst;
  asm volatile("global_load_async_to_lds_b128 %0, %1, off"
               :: "v"(loff), "v"(gsrc) : "memory");
}
__device__ __forceinline__ void wait_async0() {
  asm volatile("s_wait_asynccnt 0" ::: "memory");
}

// Load a 16x32 bf16 operand tile (row-major, leading dim ld) into the
// CDNA5 WMMA A/B lane layout (ISA 7.12.2): lane&15 = row,
// elements 0..7 = K[kb..kb+7], elements 8..15 = K[kb+16..kb+23], kb = 8*(lane>=16).
__device__ __forceinline__ v16bf frag16x32(const __bf16* p, int ld, int lane) {
  const int row = lane & 15;
  const int kb  = (lane & 16) ? 8 : 0;
  const __bf16* r0 = p + (size_t)row * ld + kb;
  v16bf f;
  *(v8bf*)&f         = *(const v8bf*)(r0);
  *(((v8bf*)&f) + 1) = *(const v8bf*)(r0 + 16);
  return f;
}

// ---------------------------------------------------------------------------
// Generic bf16 WMMA GEMM:  C[M,N] = act(A[M,K] * Bt[N,K]^T + bias) + resid
// A, Bt are padded (rows to 128, K to 32) so tile loads are unguarded.
// Double-buffered LDS tiles filled with async global->LDS copies.
// act: 0 none, 1 relu, 2 exact gelu, 3 leaky(0.01)
// ---------------------------------------------------------------------------
__global__ __launch_bounds__(256) void gemm_bf16_kernel(
    const __bf16* __restrict__ A, int lda,
    const __bf16* __restrict__ Bt, int ldb,
    const float* __restrict__ bias,
    const float* __restrict__ resid, int ldr,
    float* __restrict__ Cf, int ldc,
    __bf16* __restrict__ Cb, int ldcb,
    int M, int N, int K, int act)
{
  __shared__ alignas(16) __bf16 As[2][BM * LDSK];
  __shared__ alignas(16) __bf16 Bs[2][BN * LDSK];
  const int tid  = threadIdx.x;
  const int lane = tid & 31;
  const int wid  = tid >> 5;
  const int wm   = wid >> 2;   // 0..1  (64 rows each)
  const int wn   = wid & 3;    // 0..3  (32 cols each)
  const size_t m0 = (size_t)blockIdx.y * BM;
  const size_t n0 = (size_t)blockIdx.x * BN;

  const v8f vzero = {0.f,0.f,0.f,0.f,0.f,0.f,0.f,0.f};
  v8f acc[4][2];
#pragma unroll
  for (int i = 0; i < 4; ++i)
#pragma unroll
    for (int j = 0; j < 2; ++j) acc[i][j] = vzero;

  const int nk = K / BK;

  // Stage one 128x32 bf16 tile pair into LDS buffer `buf` (async, 4 x 16B/lane).
  auto stage = [&](int kt, int buf) {
    const int k0 = kt * BK;
#pragma unroll
    for (int it = tid; it < 512; it += 256) {
      const int row = it >> 2;
      const int c   = (it & 3) * 8;
      async_copy_b128(&As[buf][row * LDSK + c], &A [(m0 + row) * lda + k0 + c]);
      async_copy_b128(&Bs[buf][row * LDSK + c], &Bt[(n0 + row) * ldb + k0 + c]);
    }
  };

  stage(0, 0);
  wait_async0();
  __syncthreads();

  for (int kt = 0; kt < nk; ++kt) {
    const int cur = kt & 1;
    if (kt + 1 < nk) stage(kt + 1, cur ^ 1);  // overlap next-tile copy with WMMA

    v16bf bf0 = frag16x32(&Bs[cur][(wn * 32 +  0) * LDSK], LDSK, lane);
    v16bf bf1 = frag16x32(&Bs[cur][(wn * 32 + 16) * LDSK], LDSK, lane);
#pragma unroll
    for (int i = 0; i < 4; ++i) {
      v16bf af = frag16x32(&As[cur][(wm * 64 + i * 16) * LDSK], LDSK, lane);
      acc[i][0] = wmma_bf16(af, bf0, acc[i][0]);
      acc[i][1] = wmma_bf16(af, bf1, acc[i][1]);
    }

    if (kt + 1 < nk) {
      wait_async0();     // our async copies into buf^1 are complete
      __syncthreads();   // everyone's copies visible; buf^1 ready, cur reusable
    }
  }

  // Epilogue. C layout: lane&15 = N, VGPR r -> M = r + 8*(lane>=16).
  const int lrow = lane & 15;
  const int lhi  = (lane >> 4) * 8;
#pragma unroll
  for (int i = 0; i < 4; ++i)
#pragma unroll
    for (int j = 0; j < 2; ++j) {
      const int nn = (int)n0 + wn * 32 + j * 16 + lrow;
      if (nn >= N) continue;
      const float bv = bias ? bias[nn] : 0.f;
#pragma unroll
      for (int r = 0; r < 8; ++r) {
        const int mm = (int)m0 + wm * 64 + i * 16 + r + lhi;
        if (mm >= M) continue;
        float v = acc[i][j][r] + bv;
        if      (act == 1) v = fmaxf(v, 0.f);
        else if (act == 2) v = 0.5f * v * (1.f + erff(v * 0.70710678118654752f));
        else if (act == 3) v = (v > 0.f) ? v : 0.01f * v;
        if (resid) v += resid[(size_t)mm * ldr + nn];
        if (Cf) Cf[(size_t)mm * ldc  + nn] = v;
        if (Cb) Cb[(size_t)mm * ldcb + nn] = (__bf16)v;
      }
    }
}

// ---------------------------------------------------------------------------
// Fused attention: one wave per (b, h, 16 query rows). Q pre-scaled, K padded
// to 96; scores (16x256) in LDS with fused tree-bias; softmax; P*V via WMMA.
// ---------------------------------------------------------------------------
__global__ __launch_bounds__(32) void attn_kernel(
    const __bf16* __restrict__ Qb,   // [B,H,256,96] (scaled, zero pad d>=65)
    const __bf16* __restrict__ Kb,   // [B,H,256,96]
    const __bf16* __restrict__ Vt,   // [B,H,96,256] (rows d>=65 zero)
    const float* __restrict__ attn_bias,  // [B,256,256]
    const int*   __restrict__ rel_pos,    // [B,255,255]
    const float* __restrict__ rpe,        // [64,8]
    const float* __restrict__ svd,        // [8]
    __bf16* __restrict__ Obf)        // [B*256, 544] at col h*65+d
{
  __shared__ alignas(16) float  sc[16 * 256];
  __shared__ alignas(16) __bf16 pb[16 * 256];
  const int blk  = blockIdx.x;
  const int qt   = blk & 15;
  const int h    = (blk >> 4) & 7;
  const int b    = blk >> 7;
  const int lane = threadIdx.x;
  const int q0   = qt * 16;
  const int lrow = lane & 15;
  const int lhi  = (lane >> 4) * 8;

  const __bf16* Qbase = Qb + ((size_t)(b * 8 + h) * 256 + q0) * 96;
  const __bf16* Kbase = Kb + (size_t)(b * 8 + h) * 256 * 96;
  const __bf16* Vbase = Vt + (size_t)(b * 8 + h) * 96 * 256;
  const v8f vzero = {0.f,0.f,0.f,0.f,0.f,0.f,0.f,0.f};

  v16bf aq[3];
#pragma unroll
  for (int kt = 0; kt < 3; ++kt) aq[kt] = frag16x32(Qbase + kt * 32, 96, lane);

  // scores = Q*K^T (scale folded into Q) + fused bias
  for (int nt = 0; nt < 16; ++nt) {
    v8f acc = vzero;
#pragma unroll
    for (int kt = 0; kt < 3; ++kt) {
      v16bf bk = frag16x32(Kbase + (size_t)(nt * 16) * 96 + kt * 32, 96, lane);
      acc = wmma_bf16(aq[kt], bk, acc);
    }
    const int ks = nt * 16 + lrow;
#pragma unroll
    for (int r = 0; r < 8; ++r) {
      const int qs = q0 + r + lhi;
      float v = acc[r] + attn_bias[((size_t)b * 256 + qs) * 256 + ks];
      if (qs > 0 && ks > 0)
        v += rpe[(size_t)rel_pos[((size_t)b * 255 + (qs - 1)) * 255 + (ks - 1)] * 8 + h];
      if (qs == 0) v += svd[h];
      else if (ks == 0) v += svd[h];
      sc[(r + lhi) * 256 + ks] = v;
    }
  }
  __syncthreads();
  // softmax over 256 keys, one row per lane (lanes 0..15)
  if (lane < 16) {
    float mx = -3.4e38f;
    for (int k = 0; k < 256; ++k) mx = fmaxf(mx, sc[lane * 256 + k]);
    float s = 0.f;
    for (int k = 0; k < 256; ++k) {
      float e = __expf(sc[lane * 256 + k] - mx);
      sc[lane * 256 + k] = e; s += e;
    }
    const float inv = 1.f / s;
    for (int k = 0; k < 256; ++k)
      pb[lane * 256 + k] = (__bf16)(sc[lane * 256 + k] * inv);
  }
  __syncthreads();
  // O = P * V  (need d < 65; dn tiles 0..4 cover d 0..79)
  for (int dn = 0; dn < 5; ++dn) {
    v8f acc = vzero;
#pragma unroll
    for (int kt = 0; kt < 8; ++kt) {
      v16bf ap = frag16x32(pb + kt * 32, 256, lane);
      v16bf bv = frag16x32(Vbase + (size_t)(dn * 16) * 256 + kt * 32, 256, lane);
      acc = wmma_bf16(ap, bv, acc);
    }
    const int di = dn * 16 + lrow;
    if (di < 65) {
#pragma unroll
      for (int r = 0; r < 8; ++r) {
        const int qs = q0 + r + lhi;
        Obf[((size_t)b * 256 + qs) * 544 + h * 65 + di] = (__bf16)acc[r];
      }
    }
  }
}

// ---------------------------------------------------------------------------
// LayerNorm over C=521, output bf16 (padded row stride ldy).
// ---------------------------------------------------------------------------
__global__ __launch_bounds__(256) void ln_kernel(
    const float* __restrict__ x, const float* __restrict__ g,
    const float* __restrict__ bb, __bf16* __restrict__ y,
    int C, int ldx, int ldy)
{
  __shared__ float red[256];
  const int row = blockIdx.x, tid = threadIdx.x;
  const float* xr = x + (size_t)row * ldx;
  float s = 0.f, s2 = 0.f;
  for (int c = tid; c < C; c += 256) { float v = xr[c]; s += v; s2 += v * v; }
  red[tid] = s; __syncthreads();
  for (int o = 128; o > 0; o >>= 1) { if (tid < o) red[tid] += red[tid + o]; __syncthreads(); }
  const float mean = red[0] / C;
  __syncthreads();
  red[tid] = s2; __syncthreads();
  for (int o = 128; o > 0; o >>= 1) { if (tid < o) red[tid] += red[tid + o]; __syncthreads(); }
  const float var = red[0] / C - mean * mean;
  const float inv = rsqrtf(var + 1e-5f);
  __bf16* yr = y + (size_t)row * ldy;
  for (int c = tid; c < C; c += 256)
    yr[c] = (__bf16)((xr[c] - mean) * inv * g[c] + bb[c]);
}

// Weight fp32 [N,K] -> bf16 padded [Np,Kp] (pads pre-zeroed by memset)
__global__ __launch_bounds__(256) void cvt_pad_kernel(
    const float* __restrict__ W, __bf16* __restrict__ Wb, int K, int Kp)
{
  const int n = blockIdx.x;
  for (int c = threadIdx.x; c < K; c += 256)
    Wb[(size_t)n * Kp + c] = (__bf16)W[(size_t)n * K + c];
}

// Embedding lookups + lff relu -> concat256 bf16 rows (padded to 16384 rows)
__global__ __launch_bounds__(64) void embed_concat_kernel(
    const float* __restrict__ x, const float* __restrict__ te,
    const float* __restrict__ je, const float* __restrict__ tbe,
    const float* __restrict__ lffW, const float* __restrict__ lffb,
    __bf16* __restrict__ cat)
{
  const int row = blockIdx.x;       // 0..16319
  const int j   = threadIdx.x;      // 0..63
  const float* xr = x + (size_t)row * 1161;
  const int i0 = (int)xr[0], i1 = (int)xr[1], i2 = (int)xr[2];
  const float f0 = xr[1158], f1 = xr[1159], f2 = xr[1160];
  float ffe = lffW[j * 3 + 0] * f0 + lffW[j * 3 + 1] * f1 + lffW[j * 3 + 2] * f2 + lffb[j];
  ffe = fmaxf(ffe, 0.f);
  __bf16* cr = cat + (size_t)row * 256;
  cr[j]       = (__bf16)te [i0 * 64 + j];
  cr[64 + j]  = (__bf16)je [i1 * 64 + j];
  cr[128 + j] = (__bf16)tbe[i2 * 64 + j];
  cr[192 + j] = (__bf16)ffe;
}

// X[b,0,:]=super ; X[b,1+n,:]=Pf[b*255+n,:]+height_emb[heights[b,n],:]
__global__ __launch_bounds__(256) void assemble_x_kernel(
    const float* __restrict__ Pf, const float* __restrict__ hemb,
    const int* __restrict__ heights, const float* __restrict__ super_tok,
    float* __restrict__ X)
{
  const int row = blockIdx.x;  // b*256+s
  const int b = row >> 8, s = row & 255;
  float* xr = X + (size_t)row * 521;
  if (s == 0) {
    for (int c = threadIdx.x; c < 521; c += 256) xr[c] = super_tok[c];
  } else {
    const int n = s - 1;
    const int hh = heights[(size_t)b * 255 + n];
    const float* pr = Pf + (size_t)(b * 255 + n) * 521;
    const float* hr = hemb + (size_t)hh * 521;
    for (int c = threadIdx.x; c < 521; c += 256) xr[c] = pr[c] + hr[c];
  }
}

// fp32 Q/K/V [16384,520] -> bf16 Qb,Kb [B,H,256,96] (Q scaled) and Vt [B,H,96,256]
__global__ __launch_bounds__(544) void qkv_convert_kernel(
    const float* __restrict__ Qf, const float* __restrict__ Kf,
    const float* __restrict__ Vf,
    __bf16* __restrict__ Qb, __bf16* __restrict__ Kb, __bf16* __restrict__ Vt)
{
  const int row = blockIdx.x;  // b*256+s
  const int t = threadIdx.x;
  if (t >= 520) return;
  const int b = row >> 8, s = row & 255;
  const int h = t / 65, d = t % 65;
  const size_t src = (size_t)row * 520 + t;
  const size_t bh = (size_t)(b * 8 + h);
  Qb[(bh * 256 + s) * 96 + d] = (__bf16)(Qf[src] * SCALE_DK);
  Kb[(bh * 256 + s) * 96 + d] = (__bf16)(Kf[src]);
  Vt[(bh * 96 + d) * 256 + s] = (__bf16)(Vf[src]);
}

// Gather final-LN row s==0 per batch into Pa [128,544]
__global__ __launch_bounds__(544) void gather_pred_kernel(
    const __bf16* __restrict__ Ybf, __bf16* __restrict__ Pa)
{
  Pa[(size_t)blockIdx.x * 544 + threadIdx.x] =
      Ybf[(size_t)blockIdx.x * 256 * 544 + threadIdx.x];
}

// pred[b] = sigmoid(dot(Hsum[b,:], Wout) + bout)
__global__ __launch_bounds__(256) void pred_out_kernel(
    const float* __restrict__ Hsum, const float* __restrict__ W,
    const float* __restrict__ bO, float* __restrict__ out)
{
  __shared__ float red[256];
  const int b = blockIdx.x, tid = threadIdx.x;
  red[tid] = Hsum[(size_t)b * 256 + tid] * W[tid];
  __syncthreads();
  for (int o = 128; o > 0; o >>= 1) { if (tid < o) red[tid] += red[tid + o]; __syncthreads(); }
  if (tid == 0) out[b] = 1.f / (1.f + __expf(-(red[0] + bO[0])));
}

// ---------------------------------------------------------------------------
// Host
// ---------------------------------------------------------------------------
extern "C" void kernel_launch(void* const* d_in, const int* in_sizes, int n_in,
                              void* d_out, int out_size, void* d_ws, size_t ws_size,
                              hipStream_t stream)
{
  (void)in_sizes; (void)n_in; (void)out_size;
  // Input flattening: setup_inputs() dict insertion order, recursive:
  // 0 attn_bias, 1 rel_pos, 2 x, 3 heights, then params:
  // 4 type_emb, 5 join_emb, 6 table_emb, 7 lff.W, 8 lff.b, 9 project.W,
  // 10 project.b, 11 rel_pos_emb, 12 height_emb, 13 super_token, 14 svd,
  // 15 final_ln.g, 16 final_ln.b, 17..24 pred (mlp1 W/b, mid1 W/b, mid2 W/b,
  // out W/b), 25+16L layer L: ln1 g/b, q W/b, k W/b, v W/b, o W/b, ln2 g/b,
  // f1 W/b, f2 W/b.
  auto F = [&](int i) { return (const float*)d_in[i]; };
  auto Ip = [&](int i) { return (const int*)d_in[i]; };
  const float* attn_bias = F(0);
  const int*   rel_pos   = Ip(1);
  const float* xin       = F(2);
  const int*   heights   = Ip(3);

  // Workspace bump allocator (256B aligned)
  char* base = (char*)d_ws;
  size_t cur = 0;
  auto alloc = [&](size_t bytes) -> char* {
    cur = (cur + 255) & ~(size_t)255;
    char* p = base + cur; cur += bytes; return p;
  };

  __bf16 *WqB[8], *WkB[8], *WvB[8], *WoB[8], *Wf1B[8], *Wf2B[8];
  for (int L = 0; L < 8; ++L) {
    WqB[L]  = (__bf16*)alloc((size_t)640 * 544 * 2);
    WkB[L]  = (__bf16*)alloc((size_t)640 * 544 * 2);
    WvB[L]  = (__bf16*)alloc((size_t)640 * 544 * 2);
    WoB[L]  = (__bf16*)alloc((size_t)640 * 544 * 2);
    Wf1B[L] = (__bf16*)alloc((size_t)2176 * 544 * 2);
    Wf2B[L] = (__bf16*)alloc((size_t)640 * 2112 * 2);
  }
  __bf16* WprojB = (__bf16*)alloc((size_t)640 * 256 * 2);
  __bf16* Wm1B   = (__bf16*)alloc((size_t)256 * 544 * 2);
  __bf16* Wmid1B = (__bf16*)alloc((size_t)256 * 256 * 2);
  __bf16* Wmid2B = (__bf16*)alloc((size_t)256 * 256 * 2);
  __bf16* Concat = (__bf16*)alloc((size_t)16384 * 256 * 2);
  __bf16* Ybf    = (__bf16*)alloc((size_t)16384 * 544 * 2);
  float*  X      = (float*)alloc((size_t)16384 * 521 * 4);
  float*  QfBuf  = (float*)alloc((size_t)16384 * 521 * 4); // Pf (ld 521) / Qf (ld 520)
  float*  Kf     = (float*)alloc((size_t)16384 * 520 * 4);
  float*  Vf     = (float*)alloc((size_t)16384 * 520 * 4);
  __bf16* Qb     = (__bf16*)alloc((size_t)64 * 8 * 256 * 96 * 2);
  __bf16* Kb     = (__bf16*)alloc((size_t)64 * 8 * 256 * 96 * 2);
  __bf16* Vt     = (__bf16*)alloc((size_t)64 * 8 * 96 * 256 * 2);
  __bf16* Obf    = (__bf16*)alloc((size_t)16384 * 544 * 2);
  __bf16* Hb     = (__bf16*)alloc((size_t)16384 * 2112 * 2);
  __bf16* Pa     = (__bf16*)alloc((size_t)128 * 544 * 2);
  float*  Hpred  = (float*)alloc((size_t)128 * 256 * 4);
  __bf16* Hpredb = (__bf16*)alloc((size_t)128 * 256 * 2);
  __bf16* M1b    = (__bf16*)alloc((size_t)128 * 256 * 2);
  float*  Hsum   = (float*)alloc((size_t)128 * 256 * 4);

  size_t total = cur; if (total > ws_size) total = ws_size;
  hipMemsetAsync(d_ws, 0, total, stream);  // zero pads (graph-capture safe)

  auto cvt = [&](const float* W, __bf16* Wb, int N, int K, int Kp) {
    cvt_pad_kernel<<<N, 256, 0, stream>>>(W, Wb, K, Kp);
  };
  for (int L = 0; L < 8; ++L) {
    const int p = 25 + 16 * L;
    cvt(F(p + 2),  WqB[L],  520, 521, 544);
    cvt(F(p + 4),  WkB[L],  520, 521, 544);
    cvt(F(p + 6),  WvB[L],  520, 521, 544);
    cvt(F(p + 8),  WoB[L],  521, 520, 544);
    cvt(F(p + 12), Wf1B[L], 2084, 521, 544);
    cvt(F(p + 14), Wf2B[L], 521, 2084, 2112);
  }
  cvt(F(9),  WprojB, 521, 256, 256);
  cvt(F(17), Wm1B,   256, 521, 544);
  cvt(F(19), Wmid1B, 256, 256, 256);
  cvt(F(21), Wmid2B, 256, 256, 256);

  auto gemm = [&](const __bf16* A, int lda, const __bf16* Bt, int ldb,
                  const float* bias, const float* resid, int ldr,
                  float* Cf, int ldc, __bf16* Cb, int ldcb,
                  int M, int Mpad, int N, int Np, int K, int act) {
    dim3 g(Np / 128, Mpad / 128);
    gemm_bf16_kernel<<<g, 256, 0, stream>>>(A, lda, Bt, ldb, bias, resid, ldr,
                                            Cf, ldc, Cb, ldcb, M, N, K, act);
  };

  // Front-end: embeddings -> concat256 -> project (leaky) -> X assembly
  embed_concat_kernel<<<16320, 64, 0, stream>>>(xin, F(4), F(5), F(6), F(7), F(8), Concat);
  gemm(Concat, 256, WprojB, 256, F(10), nullptr, 0, QfBuf, 521, nullptr, 0,
       16320, 16384, 521, 640, 256, 3);
  assemble_x_kernel<<<16384, 256, 0, stream>>>(QfBuf, F(12), heights, F(13), X);

  auto run_layer = [&](int L) {
    const int p = 25 + 16 * L;
    ln_kernel<<<16384, 256, 0, stream>>>(X, F(p + 0), F(p + 1), Ybf, 521, 521, 544);
    gemm(Ybf, 544, WqB[L], 544, F(p + 3), nullptr, 0, QfBuf, 520, nullptr, 0,
         16384, 16384, 520, 640, 544, 0);
    gemm(Ybf, 544, WkB[L], 544, F(p + 5), nullptr, 0, Kf, 520, nullptr, 0,
         16384, 16384, 520, 640, 544, 0);
    gemm(Ybf, 544, WvB[L], 544, F(p + 7), nullptr, 0, Vf, 520, nullptr, 0,
         16384, 16384, 520, 640, 544, 0);
    qkv_convert_kernel<<<16384, 544, 0, stream>>>(QfBuf, Kf, Vf, Qb, Kb, Vt);
    attn_kernel<<<8192, 32, 0, stream>>>(Qb, Kb, Vt, attn_bias, rel_pos,
                                         F(11), F(14), Obf);
    gemm(Obf, 544, WoB[L], 544, F(p + 9), X, 521, X, 521, nullptr, 0,
         16384, 16384, 521, 640, 544, 0);
    ln_kernel<<<16384, 256, 0, stream>>>(X, F(p + 10), F(p + 11), Ybf, 521, 521, 544);
    gemm(Ybf, 544, Wf1B[L], 544, F(p + 13), nullptr, 0, nullptr, 0, Hb, 2112,
         16384, 16384, 2084, 2176, 544, 2);
    gemm(Hb, 2112, Wf2B[L], 2112, F(p + 15), X, 521, X, 521, nullptr, 0,
         16384, 16384, 521, 640, 2112, 0);
  };

  for (int L = 0; L < 7; ++L) run_layer(L);
  hipMemcpyAsync((float*)d_out + 64, X, (size_t)16384 * 521 * sizeof(float),
                 hipMemcpyDeviceToDevice, stream);   // second_to_last
  run_layer(7);

  // Final LN + prediction head
  ln_kernel<<<16384, 256, 0, stream>>>(X, F(15), F(16), Ybf, 521, 521, 544);
  gather_pred_kernel<<<64, 544, 0, stream>>>(Ybf, Pa);
  gemm(Pa, 544, Wm1B, 544, F(18), nullptr, 0, Hpred, 256, Hpredb, 256,
       64, 128, 256, 256, 544, 1);
  gemm(Hpredb, 256, Wmid1B, 256, F(20), nullptr, 0, nullptr, 0, M1b, 256,
       64, 128, 256, 256, 256, 1);
  gemm(M1b, 256, Wmid2B, 256, F(22), Hpred, 256, Hsum, 256, nullptr, 0,
       64, 128, 256, 256, 256, 1);
  pred_out_kernel<<<64, 256, 0, stream>>>(Hsum, F(23), F(24), (float*)d_out);
}
